// QuantLlamaMLP_47339129536591
// MI455X (gfx1250) — compile-verified, matrix operand order checked by reading, not run on hardware
//
#include <hip/hip_runtime.h>
#include <cstdint>

#define MDIM 4096   // B*S
#define KDIM 4096
#define NDIM 11008
#define GS   128

#if defined(__HIP_DEVICE_COMPILE__) && \
    __has_builtin(__builtin_amdgcn_tensor_load_to_lds) && \
    __has_builtin(__builtin_amdgcn_s_wait_tensorcnt)
#define USE_TDM 1
#else
#define USE_TDM 0
#endif

typedef __attribute__((ext_vector_type(16))) _Float16 v16h;
typedef __attribute__((ext_vector_type(8)))  _Float16 v8h;
typedef __attribute__((ext_vector_type(4)))  _Float16 v4h;
typedef __attribute__((ext_vector_type(8)))  float    v8f;
typedef __attribute__((ext_vector_type(4)))  unsigned int u32x4;
#if USE_TDM
typedef __attribute__((ext_vector_type(8)))  int i32x8;
typedef __attribute__((ext_vector_type(4)))  int i32x4;
#endif

union Frag {
    v16h  h;
    u32x4 q[2];
};

#if USE_TDM
// ---------------------------------------------------------------------------
// Tensor Data Mover: async 2D tile load (d1 rows of d0 2-byte elements,
// row stride stride0 elements) from global into LDS at byte offset ldsDst.
// Descriptor bit layout per CDNA5 ISA.
// 6-arg builtin form (clang-23 / therock-10.0 lane).
// ---------------------------------------------------------------------------
__device__ __forceinline__ void tdm_load_2d_f16(const void* gsrc, unsigned ldsDst,
                                                unsigned d0, unsigned d1,
                                                unsigned stride0) {
    unsigned long long ga = (unsigned long long)(uintptr_t)gsrc;
    u32x4 g0;
    g0[0] = 1u;                                    // count=1 (valid), user mode
    g0[1] = ldsDst;                                // lds_addr[31:0]
    g0[2] = (unsigned)ga;                          // global_addr[31:0]
    g0[3] = ((unsigned)(ga >> 32) & 0x01ffffffu)   // global_addr[56:32]
            | (2u << 30);                          // type=2 ("image")
    i32x8 g1;
    g1[0] = (int)(1u << 16);                       // wg_mask=0, data_size=1 (2B)
    g1[1] = (int)((d0 & 0xffffu) << 16);           // tensor_dim0[15:0] @ bits[63:48]
    g1[2] = (int)(((d0 >> 16) & 0xffffu) |         // tensor_dim0[31:16]
                  ((d1 & 0xffffu) << 16));         // tensor_dim1[15:0]
    g1[3] = (int)(((d1 >> 16) & 0xffffu) |         // tensor_dim1[31:16]
                  ((d0 & 0xffffu) << 16));         // tile_dim0 = d0
    g1[4] = (int)(d1 & 0xffffu);                   // tile_dim1 = d1, tile_dim2=0
    g1[5] = (int)stride0;                          // tensor_dim0_stride[31:0]
    g1[6] = 0;                                     // stride hi, dim1_stride lo
    g1[7] = 0;
    i32x4 gz4 = {0, 0, 0, 0};
    i32x8 gz8 = {0, 0, 0, 0, 0, 0, 0, 0};
    __builtin_amdgcn_tensor_load_to_lds(g0, g1, gz4, gz4, gz8, 0);
}
#endif

// ---------------------------------------------------------------------------
// Kernel 1: cast x (fp32) -> f16, 4 elements per thread
// ---------------------------------------------------------------------------
__global__ __launch_bounds__(256)
void cvt_x_f16(const float* __restrict__ x, _Float16* __restrict__ xh) {
    int i = (blockIdx.x * 256 + threadIdx.x) * 4;
    float4 v = *(const float4*)(x + i);
    v4h o;
    o[0] = (_Float16)v.x; o[1] = (_Float16)v.y;
    o[2] = (_Float16)v.z; o[3] = (_Float16)v.w;
    *(v4h*)(xh + i) = o;
}

// ---------------------------------------------------------------------------
// Kernel 2: fused gate/up GEMM + SiLU.  h = silu(x@Wg) * (x@Wu)  -> f16
// Block tile 128Mx64N, K-step 32, double-buffered LDS software pipeline.
// A tile staged by TDM (wave 0); weight tiles dequantized int4->f16 in VALU.
// 8 waves (4Mx2N); each wave 32x32 -> 2x2 WMMA tiles for gate AND up.
// ---------------------------------------------------------------------------
__global__ __launch_bounds__(256)
void gateup_silu_kernel(const _Float16* __restrict__ xh,
                        const int*   __restrict__ gq,
                        const float* __restrict__ gsc,
                        const int*   __restrict__ gz,
                        const int*   __restrict__ uq,
                        const float* __restrict__ usc,
                        const int*   __restrict__ uz,
                        _Float16* __restrict__ hbuf) {
    __shared__ __align__(16) _Float16 sA[2][128 * 32];  // [m][k]
    __shared__ __align__(16) _Float16 sG[2][64 * 32];   // [n][k]
    __shared__ __align__(16) _Float16 sU[2][64 * 32];   // [n][k]

    const int tid  = threadIdx.x;
    const int bm   = blockIdx.y * 128;
    const int bn   = blockIdx.x * 64;
    const int wave = tid >> 5;
    const int lane = tid & 31;
    const int wm   = wave & 3;
    const int wn   = wave >> 2;
    const int hrow = lane & 15;
    const int khalf = (lane >> 4) * 8;
#if !USE_TDM
    const int arow = tid >> 1;
    const int aseg = (tid & 1) * 16;
#endif
    const int col  = tid & 63;
    const int oct  = tid >> 6;
    const int ncol = bn + col;

    v8f accG[2][2], accU[2][2];
#pragma unroll
    for (int i = 0; i < 2; ++i)
#pragma unroll
        for (int j = 0; j < 2; ++j) { accG[i][j] = (v8f)0.0f; accU[i][j] = (v8f)0.0f; }

    auto stage = [&](int b, int k0) {
#if USE_TDM
        if (wave == 0)
            tdm_load_2d_f16(&xh[(size_t)bm * KDIM + k0],
                            (unsigned)(uintptr_t)&sA[b][0], 32u, 128u, (unsigned)KDIM);
#else
        *(u32x4*)&sA[b][arow * 32 + aseg] =
            *(const u32x4*)&xh[(bm + arow) * KDIM + k0 + aseg];
#endif
        const int grp = k0 / GS;
        {
            int   q  = gq[(k0 / 8 + oct) * NDIM + ncol];
            float sc = gsc[grp * NDIM + ncol];
            int   z  = (gz[grp * (NDIM / 8) + (ncol >> 3)] >> ((ncol & 7) * 4)) & 15;
            float zs = (float)(z + 1) * sc;
            v8h vals;
#pragma unroll
            for (int j = 0; j < 8; ++j)
                vals[j] = (_Float16)((float)((q >> (4 * j)) & 15) * sc - zs);
            *(v8h*)&sG[b][col * 32 + oct * 8] = vals;
        }
        {
            int   q  = uq[(k0 / 8 + oct) * NDIM + ncol];
            float sc = usc[grp * NDIM + ncol];
            int   z  = (uz[grp * (NDIM / 8) + (ncol >> 3)] >> ((ncol & 7) * 4)) & 15;
            float zs = (float)(z + 1) * sc;
            v8h vals;
#pragma unroll
            for (int j = 0; j < 8; ++j)
                vals[j] = (_Float16)((float)((q >> (4 * j)) & 15) * sc - zs);
            *(v8h*)&sU[b][col * 32 + oct * 8] = vals;
        }
    };

    // prologue
    stage(0, 0);
#if USE_TDM
    if (wave == 0) __builtin_amdgcn_s_wait_tensorcnt(0);
#endif
    __syncthreads();

    for (int k0 = 0; k0 < KDIM; k0 += 32) {
        const int p = (k0 >> 5) & 1;
        if (k0 + 32 < KDIM) stage(p ^ 1, k0 + 32);   // overlap next tile with compute

        Frag a[2], bg[2], bu[2];
#pragma unroll
        for (int mi = 0; mi < 2; ++mi) {
            int r = wm * 32 + mi * 16 + hrow;
            a[mi].q[0] = *(const u32x4*)&sA[p][r * 32 + khalf];
            a[mi].q[1] = *(const u32x4*)&sA[p][r * 32 + 16 + khalf];
        }
#pragma unroll
        for (int ni = 0; ni < 2; ++ni) {
            int c = wn * 32 + ni * 16 + hrow;
            bg[ni].q[0] = *(const u32x4*)&sG[p][c * 32 + khalf];
            bg[ni].q[1] = *(const u32x4*)&sG[p][c * 32 + 16 + khalf];
            bu[ni].q[0] = *(const u32x4*)&sU[p][c * 32 + khalf];
            bu[ni].q[1] = *(const u32x4*)&sU[p][c * 32 + 16 + khalf];
        }

#pragma unroll
        for (int mi = 0; mi < 2; ++mi)
#pragma unroll
            for (int ni = 0; ni < 2; ++ni) {
                accG[mi][ni] = __builtin_amdgcn_wmma_f32_16x16x32_f16(
                    false, a[mi].h, false, bg[ni].h, (short)0, accG[mi][ni], false, false);
                accU[mi][ni] = __builtin_amdgcn_wmma_f32_16x16x32_f16(
                    false, a[mi].h, false, bu[ni].h, (short)0, accU[mi][ni], false, false);
            }

#if USE_TDM
        if (wave == 0) __builtin_amdgcn_s_wait_tensorcnt(0);
#endif
        __syncthreads();
    }

    // epilogue: h = silu(g) * u -> f16, C/D layout (VGPR e <-> row e / e+8)
    // sigmoid via hardware v_rcp_f32 instead of IEEE divide sequence
    const int rbase = bm + wm * 32 + ((lane >> 4) * 8);
    const int cbase = bn + wn * 32 + (lane & 15);
#pragma unroll
    for (int mi = 0; mi < 2; ++mi)
#pragma unroll
        for (int ni = 0; ni < 2; ++ni)
#pragma unroll
            for (int e = 0; e < 8; ++e) {
                float g = accG[mi][ni][e];
                float u = accU[mi][ni][e];
                float sig = __builtin_amdgcn_rcpf(1.0f + __expf(-g));
                float h = g * sig * u;
                hbuf[(rbase + mi * 16 + e) * NDIM + (cbase + ni * 16)] = (_Float16)h;
            }
}

// ---------------------------------------------------------------------------
// Kernel 3: down GEMM.  out = h @ Wd  (reduction over N=11008), fp32 out.
// Block tile 128Mx128K, red-step 32, double-buffered. Wave: 32x64 = 2x4 tiles.
// ---------------------------------------------------------------------------
__global__ __launch_bounds__(256)
void down_kernel(const _Float16* __restrict__ hbuf,
                 const int*   __restrict__ dq,
                 const float* __restrict__ dsc,
                 const int*   __restrict__ dz,
                 float* __restrict__ out) {
    __shared__ __align__(16) _Float16 sA[2][128 * 32];  // [m][n_red]
    __shared__ __align__(16) _Float16 sW[2][128 * 32];  // [k_out][n_red]

    const int tid  = threadIdx.x;
    const int bm   = blockIdx.y * 128;
    const int bk   = blockIdx.x * 128;
    const int wave = tid >> 5;
    const int lane = tid & 31;
    const int wm   = wave & 3;
    const int wn   = wave >> 2;
    const int hrow = lane & 15;
    const int khalf = (lane >> 4) * 8;
#if !USE_TDM
    const int arow = tid >> 1;
    const int aseg = (tid & 1) * 16;
#endif

    v8f acc[2][4];
#pragma unroll
    for (int i = 0; i < 2; ++i)
#pragma unroll
        for (int j = 0; j < 4; ++j) acc[i][j] = (v8f)0.0f;

    auto stage = [&](int b, int n0) {
#if USE_TDM
        if (wave == 0)
            tdm_load_2d_f16(&hbuf[(size_t)bm * NDIM + n0],
                            (unsigned)(uintptr_t)&sA[b][0], 32u, 128u, (unsigned)NDIM);
#else
        *(u32x4*)&sA[b][arow * 32 + aseg] =
            *(const u32x4*)&hbuf[(bm + arow) * NDIM + n0 + aseg];
#endif
        const int grp = n0 / GS;
#pragma unroll
        for (int it = 0; it < 2; ++it) {
            int idx = tid + it * 256;
            int c   = idx & 127;
            int oc  = idx >> 7;
            int kc  = bk + c;
            int   q  = dq[(n0 / 8 + oc) * KDIM + kc];
            float sc = dsc[grp * KDIM + kc];
            int   z  = (dz[grp * (KDIM / 8) + (kc >> 3)] >> ((kc & 7) * 4)) & 15;
            float zs = (float)(z + 1) * sc;
            v8h vals;
#pragma unroll
            for (int j = 0; j < 8; ++j)
                vals[j] = (_Float16)((float)((q >> (4 * j)) & 15) * sc - zs);
            *(v8h*)&sW[b][c * 32 + oc * 8] = vals;
        }
    };

    stage(0, 0);
#if USE_TDM
    if (wave == 0) __builtin_amdgcn_s_wait_tensorcnt(0);
#endif
    __syncthreads();

    for (int n0 = 0; n0 < NDIM; n0 += 32) {
        const int p = (n0 >> 5) & 1;
        if (n0 + 32 < NDIM) stage(p ^ 1, n0 + 32);

        Frag a[2], b[4];
#pragma unroll
        for (int mi = 0; mi < 2; ++mi) {
            int r = wm * 32 + mi * 16 + hrow;
            a[mi].q[0] = *(const u32x4*)&sA[p][r * 32 + khalf];
            a[mi].q[1] = *(const u32x4*)&sA[p][r * 32 + 16 + khalf];
        }
#pragma unroll
        for (int ni = 0; ni < 4; ++ni) {
            int c = wn * 64 + ni * 16 + hrow;
            b[ni].q[0] = *(const u32x4*)&sW[p][c * 32 + khalf];
            b[ni].q[1] = *(const u32x4*)&sW[p][c * 32 + 16 + khalf];
        }
#pragma unroll
        for (int mi = 0; mi < 2; ++mi)
#pragma unroll
            for (int ni = 0; ni < 4; ++ni)
                acc[mi][ni] = __builtin_amdgcn_wmma_f32_16x16x32_f16(
                    false, a[mi].h, false, b[ni].h, (short)0, acc[mi][ni], false, false);

#if USE_TDM
        if (wave == 0) __builtin_amdgcn_s_wait_tensorcnt(0);
#endif
        __syncthreads();
    }

    const int rbase = bm + wm * 32 + ((lane >> 4) * 8);
    const int cbase = bk + wn * 64 + (lane & 15);
#pragma unroll
    for (int mi = 0; mi < 2; ++mi)
#pragma unroll
        for (int ni = 0; ni < 4; ++ni)
#pragma unroll
            for (int e = 0; e < 8; ++e)
                out[(rbase + mi * 16 + e) * KDIM + (cbase + ni * 16)] = acc[mi][ni][e];
}

// ---------------------------------------------------------------------------
extern "C" void kernel_launch(void* const* d_in, const int* in_sizes, int n_in,
                              void* d_out, int out_size, void* d_ws, size_t ws_size,
                              hipStream_t stream) {
    const float* x   = (const float*)d_in[0];
    const int*   gq  = (const int*)  d_in[1];
    const float* gsc = (const float*)d_in[2];
    const int*   gz  = (const int*)  d_in[3];
    const int*   uq  = (const int*)  d_in[4];
    const float* usc = (const float*)d_in[5];
    const int*   uz  = (const int*)  d_in[6];
    const int*   dq  = (const int*)  d_in[7];
    const float* dsc = (const float*)d_in[8];
    const int*   dz  = (const int*)  d_in[9];
    float* out = (float*)d_out;

    const int M = in_sizes[0] / KDIM;    // 4096

    _Float16* xh   = (_Float16*)d_ws;
    _Float16* hbuf = (_Float16*)((char*)d_ws + (size_t)M * KDIM * sizeof(_Float16));

    int total = M * KDIM;
    cvt_x_f16<<<total / 1024, 256, 0, stream>>>(x, xh);

    gateup_silu_kernel<<<dim3(NDIM / 64, M / 128), 256, 0, stream>>>(
        xh, gq, gsc, gz, uq, usc, uz, hbuf);

    down_kernel<<<dim3(KDIM / 128, M / 128), 256, 0, stream>>>(
        hbuf, dq, dsc, dz, out);
}
